// MMN_34995393527847
// MI455X (gfx1250) — matrix operand order repaired and checked by reading, not compiled, not run on hardware
//
#include <hip/hip_runtime.h>
#include <math.h>

// Problem sizes (fixed by the reference)
#define Bv 16
#define Sv 8
#define Nv 64
#define Dv 512
#define Jv 256
#define Kv 128

typedef __attribute__((ext_vector_type(16))) __bf16 v16bf;
typedef __attribute__((ext_vector_type(2)))  __bf16 v2bf;
typedef __attribute__((ext_vector_type(8)))  float  v8f;

struct alignas(16) U4 { unsigned x, y, z, w; };
union BV { U4 q[2]; unsigned u[8]; v16bf v; };

// pack two f32 -> two bf16 in one dword (hardware cvt if available)
__device__ __forceinline__ unsigned pk2bf(float a, float b) {
#if __has_builtin(__builtin_amdgcn_cvt_pk_bf16_f32)
  union { v2bf v; unsigned u; } c;
  c.v = __builtin_amdgcn_cvt_pk_bf16_f32(a, b);
  return c.u;
#else
  union { float f; unsigned u; } x, y; x.f = a; y.f = b;
  return (y.u & 0xFFFF0000u) | (x.u >> 16);     // truncation pack
#endif
}

__device__ __forceinline__ float sigmoid10(float v) {
  return 1.f / (1.f + __expf(-10.f * v));
}

// ---------------------------------------------------------------------------
// Kernel 0: pack Wcomb = [Wiou ; Wp] (f32 [512][512]) into bf16 laid out
// exactly as per-lane WMMA A fragments, so the GEMM loads A with two
// coalesced global b128 reads and zero conversion VALU in the hot loop.
// ---------------------------------------------------------------------------
__global__ void __launch_bounds__(256) mmn_packw_kernel(
    const float* __restrict__ Wiou, const float* __restrict__ Wp,
    __bf16* __restrict__ Apk)
{
  int p = blockIdx.x * 256 + threadIdx.x;          // pair index, 131072 total
  if (p >= (512 * 512) / 2) return;
  int j  = p >> 8;                                 // 256 pairs per row
  int kp = (p & 255) * 2;
  const float* src =
      (j < Jv ? Wiou + (size_t)j * Dv : Wp + (size_t)(j - Jv) * Dv) + kp;
  unsigned pk = pk2bf(src[0], src[1]);
  int jt = j >> 4, l16 = j & 15;
  int ks = kp >> 5, ko = kp & 31;
  int blk  = ko >> 3;                              // 0..3 (8-wide K blocks)
  int lane = l16 + ((blk & 1) ? 16 : 0);
  int h    = (ko & 7) + ((blk >= 2) ? 8 : 0);
  size_t off = ((size_t)((jt * 16 + ks) * 32 + lane)) * 16 + h;
  *(unsigned*)(Apk + off) = pk;                    // h is even -> dword aligned
}

// ---------------------------------------------------------------------------
// Kernel 1: per (b, n) workgroup.
//   C[j(512 = iou | contrastive), m(64)] = Wcomb @ (f[b,:,n]*f[b,:,m])
//   via v_wmma_f32_16x16x32_bf16; j-reduction (norm + 8 sentence dots/head)
//   straight from accumulators, merged with ds_add_f32; planes 0 and 2.
// ---------------------------------------------------------------------------
template <bool USE_PK>
__global__ void __launch_bounds__(256) mmn_map2d_kernel(
    const float* __restrict__ feats, const float* __restrict__ sent_feat,
    const float* __restrict__ sent_iou, const float* __restrict__ Wp,
    const float* __restrict__ Wiou, const __bf16* __restrict__ Apk,
    float* __restrict__ out)
{
  extern __shared__ char smem[];
  const int RS = 520;                                    // halves/row of Bl
  __bf16* Bl    = (__bf16*)smem;                         // 66560 B
  float*  nsent = (float*)(smem + 66560);                // 16384 B
  float*  fn    = (float*)(smem + 66560 + 16384);        //  2048 B
  float*  part  = (float*)(smem + 66560 + 16384 + 2048); //  4608 B

  const int tid  = threadIdx.x;
  const int n    = blockIdx.x;
  const int b    = blockIdx.y;
  const int lane = tid & 31;
  const int w    = tid >> 5;
  const bool hi  = lane >= 16;
  const int l16  = lane & 15;
  const int head = w >> 2;                   // 0 = iou, 1 = contrastive
  const int jbase = (w & 3) * 64;            // row base within head

  // ---- phase 0: stage f[b,:,n], zero partials, normalized sentence feats
  {
    const float* fb = feats + (size_t)b * Dv * Nv;
    for (int d = tid; d < Dv; d += 256) fn[d] = fb[d * Nv + n];
    for (int i = tid; i < 2 * 9 * 64; i += 256) part[i] = 0.f;
    if (tid < 16) {
      int hh = tid >> 3, s = tid & 7;
      const float* src = (hh == 0 ? sent_iou : sent_feat) + ((size_t)b * Sv + s) * Jv;
      float ss = 0.f;
      for (int j = 0; j < Jv; ++j) { float v = src[j]; ss += v * v; }
      float inv = 1.f / fmaxf(sqrtf(ss), 1e-12f);
      float* dst = nsent + tid * Jv;
      for (int j = 0; j < Jv; ++j) dst[j] = src[j] * inv;
    }
  }
  __syncthreads();

  // ---- phase 1: stage B operand Bl[m][d] = f[b,d,m] * f[b,d,n]  (bf16)
  {
    const float* fb = feats + (size_t)b * Dv * Nv;
    int m  = tid & 63;
    int d0 = (tid >> 6) * 128;
    __bf16* row = Bl + m * RS;
    for (int d = d0; d < d0 + 128; d += 2) {
      float a0 = fb[d * Nv + m]       * fn[d];
      float a1 = fb[(d + 1) * Nv + m] * fn[d + 1];
      *(unsigned*)(row + d) = pk2bf(a0, a1);
    }
  }
  __syncthreads();

  // ---- phase 2: GEMM
  v8f zero = {};
  v8f acc[4][4];
#pragma unroll
  for (int a = 0; a < 4; ++a)
#pragma unroll
    for (int c = 0; c < 4; ++c) acc[a][c] = zero;

  const float* Wsel = (head == 0 ? Wiou : Wp);

  for (int ks = 0; ks < 16; ++ks) {
    const int kb = ks * 32;
    v16bf Bt[4];                 // B[k,m]: lane<16 K 0..7,16..23, lane>=16 K 8..15,24..31
#pragma unroll
    for (int mt = 0; mt < 4; ++mt) {
      const __bf16* p = Bl + (mt * 16 + l16) * RS + kb + (hi ? 8 : 0);
      BV u; u.q[0] = *(const U4*)p; u.q[1] = *(const U4*)(p + 16);
      Bt[mt] = u.v;
    }
#pragma unroll
    for (int jt = 0; jt < 4; ++jt) {
      v16bf A;
      if constexpr (USE_PK) {
        // pre-packed fragment: two coalesced 16B loads, zero conversion
        const U4* ap = (const U4*)Apk +
            (size_t)(((head * 16 + (w & 3) * 4 + jt) * 16 + ks) * 32 + lane) * 2;
        BV u; u.q[0] = ap[0]; u.q[1] = ap[1];
        A = u.v;
      } else {
        const float* wr = Wsel + (size_t)(jbase + jt * 16 + l16) * Dv;
        const float* p0 = wr + kb + (hi ? 8 : 0);
        const float* p1 = wr + kb + 16 + (hi ? 8 : 0);
        BV u;
#pragma unroll
        for (int i = 0; i < 4; ++i) u.u[i]     = pk2bf(p0[2*i], p0[2*i+1]);
#pragma unroll
        for (int i = 0; i < 4; ++i) u.u[i + 4] = pk2bf(p1[2*i], p1[2*i+1]);
        A = u.v;
      }
#pragma unroll
      for (int mt = 0; mt < 4; ++mt)
        acc[jt][mt] = __builtin_amdgcn_wmma_f32_16x16x32_bf16(
            false, A, false, Bt[mt], (short)0, acc[jt][mt], false, false);
    }
  }

  // ---- phase 3: j-reduction from accumulators
  // element: j = jbase + jt*16 + r + (hi?8:0); m = mt*16 + l16
  float sq[4] = {0.f, 0.f, 0.f, 0.f};
  float dt[4][8];
#pragma unroll
  for (int mt = 0; mt < 4; ++mt)
#pragma unroll
    for (int s = 0; s < 8; ++s) dt[mt][s] = 0.f;

#pragma unroll
  for (int jt = 0; jt < 4; ++jt) {
#pragma unroll
    for (int r = 0; r < 8; ++r) {
      const int jl = jbase + jt * 16 + r + (hi ? 8 : 0);
      float ns[8];
#pragma unroll
      for (int s = 0; s < 8; ++s) ns[s] = nsent[(head * 8 + s) * Jv + jl];
#pragma unroll
      for (int mt = 0; mt < 4; ++mt) {
        float a = acc[jt][mt][r];
        sq[mt] += a * a;
#pragma unroll
        for (int s = 0; s < 8; ++s) dt[mt][s] += a * ns[s];
      }
    }
  }
#pragma unroll
  for (int mt = 0; mt < 4; ++mt) {
    const int m = mt * 16 + l16;
    atomicAdd(&part[(head * 9 + 0) * 64 + m], sq[mt]);     // ds_add_f32
#pragma unroll
    for (int s = 0; s < 8; ++s)
      atomicAdd(&part[(head * 9 + 1 + s) * 64 + m], dt[mt][s]);
  }
  __syncthreads();

  // ---- phase 4: finalize, write planes 0 (iou, sigmoid) and 2 (contrastive)
  if (tid < 128) {
    const int head2 = tid >> 6, m = tid & 63;
    const float inv = 1.f / fmaxf(sqrtf(part[(head2 * 9) * 64 + m]), 1e-12f);
    const bool keep = (m >= n);
    const size_t plane = (size_t)(head2 == 0 ? 0 : 2) * Bv * Sv * Nv * Nv;
    const size_t base  = (size_t)b * Sv * Nv * Nv + (size_t)n * Nv + m;
#pragma unroll
    for (int s = 0; s < 8; ++s) {
      float v = part[(head2 * 9 + 1 + s) * 64 + m] * inv;
      float o = (head2 == 0) ? (keep ? sigmoid10(v) : 0.f)
                             : (keep ? v : 0.f);
      out[plane + base + (size_t)s * Nv * Nv] = o;
    }
  }
}

// ---------------------------------------------------------------------------
// Kernel 2: fusion branch. One workgroup per b, one wave per sentence s.
//   score = Sum_k (Wfuse[k]*q2[s,k]^2) * v1[k,n] * v1[k,m]  via WMMA
// v1 kept twice in LDS: f32 (for the per-sentence scaled A operand) and a
// bf16 mirror built once (so B tiles are two plain ds_load_b128).
// ---------------------------------------------------------------------------
__global__ void __launch_bounds__(256) mmn_fusion_kernel(
    const float* __restrict__ feats, const float* __restrict__ sent_fuse,
    const float* __restrict__ W1, const float* __restrict__ b1,
    const float* __restrict__ W2, const float* __restrict__ b2,
    const float* __restrict__ Wfuse, float* __restrict__ out)
{
  extern __shared__ char smem[];
  const int VR  = 132;                       // f32 row stride (floats)
  const int VRH = 136;                       // bf16 row stride (halves), 272B rows
  float*  q2w = (float*)smem;                              //  4096 B
  float*  v1  = (float*)(smem + 4096);                     // 33792 B
  __bf16* v1h = (__bf16*)(smem + 4096 + 33792);            // 17408 B

  const int tid  = threadIdx.x, b = blockIdx.x;
  const int lane = tid & 31,    w = tid >> 5;
  const bool hi  = lane >= 16;
  const int l16  = lane & 15;

  // q2 -> w[s,k] = Wfuse[k]*q2^2
  {
    int o = tid;
#pragma unroll
    for (int i = 0; i < 4; ++i, o += 256) {
      int s = o >> 7, k = o & 127;
      const float* sf = sent_fuse + ((size_t)b * Sv + s) * Dv;
      float acc = b2[k];
      for (int d = 0; d < Dv; ++d) acc += sf[d] * W2[d * Kv + k];
      q2w[s * Kv + k] = Wfuse[k] * acc * acc;
    }
  }
  // v1[n][k] (f32), n fastest across lanes -> coalesced feats reads
  {
    const float* fb = feats + (size_t)b * Dv * Nv;
    int o = tid;
    for (int i = 0; i < 32; ++i, o += 256) {
      int k = o >> 6, nn = o & 63;
      float acc = b1[k];
      for (int d = 0; d < Dv; ++d) acc += fb[d * Nv + nn] * W1[d * Kv + k];
      v1[nn * VR + k] = acc;
    }
  }
  __syncthreads();
  // bf16 mirror of v1, packed pairwise along k (once, shared by all waves)
  {
    int o = tid;                             // 64 rows * 64 pairs = 4096 slots
    for (int i = 0; i < 16; ++i, o += 256) {
      int nn = o >> 6, kp = (o & 63) * 2;
      const float* src = v1 + nn * VR + kp;
      *(unsigned*)(v1h + nn * VRH + kp) = pk2bf(src[0], src[1]);
    }
  }
  __syncthreads();

  const int s = w;
  const float* ws = q2w + s * Kv;
  v8f zero = {};
  v8f acc[4][4];
#pragma unroll
  for (int a = 0; a < 4; ++a)
#pragma unroll
    for (int c = 0; c < 4; ++c) acc[a][c] = zero;

  for (int ks = 0; ks < 4; ++ks) {
    const int kb = ks * 32;
    v16bf Bt[4];                 // B[k,m] from bf16 mirror: 2x ds_load_b128
#pragma unroll
    for (int mt = 0; mt < 4; ++mt) {
      const __bf16* p = v1h + (mt * 16 + l16) * VRH + kb + (hi ? 8 : 0);
      BV u; u.q[0] = *(const U4*)p; u.q[1] = *(const U4*)(p + 16);
      Bt[mt] = u.v;
    }
#pragma unroll
    for (int nt = 0; nt < 4; ++nt) {
      const float* vr = v1 + (nt * 16 + l16) * VR;     // A[n,k] = w[s,k]*v1[k,n]
      const int o0 = kb + (hi ? 8 : 0), o1 = kb + 16 + (hi ? 8 : 0);
      BV u;
#pragma unroll
      for (int i = 0; i < 4; ++i)
        u.u[i]     = pk2bf(vr[o0+2*i] * ws[o0+2*i], vr[o0+2*i+1] * ws[o0+2*i+1]);
#pragma unroll
      for (int i = 0; i < 4; ++i)
        u.u[i + 4] = pk2bf(vr[o1+2*i] * ws[o1+2*i], vr[o1+2*i+1] * ws[o1+2*i+1]);
      v16bf A = u.v;
#pragma unroll
      for (int mt = 0; mt < 4; ++mt)
        acc[nt][mt] = __builtin_amdgcn_wmma_f32_16x16x32_bf16(
            false, A, false, Bt[mt], (short)0, acc[nt][mt], false, false);
    }
  }

  const size_t base = ((size_t)(Bv + b) * Sv + s) * (Nv * Nv);
#pragma unroll
  for (int nt = 0; nt < 4; ++nt)
#pragma unroll
    for (int mt = 0; mt < 4; ++mt)
#pragma unroll
      for (int r = 0; r < 8; ++r) {
        const int nn = nt * 16 + r + (hi ? 8 : 0);
        const int m  = mt * 16 + l16;
        float v = acc[nt][mt][r];
        out[base + (size_t)nn * Nv + m] = (m >= nn) ? sigmoid10(v) : 0.f;
      }
}

// ---------------------------------------------------------------------------
extern "C" void kernel_launch(void* const* d_in, const int* in_sizes, int n_in,
                              void* d_out, int out_size, void* d_ws, size_t ws_size,
                              hipStream_t stream) {
  (void)in_sizes; (void)n_in; (void)out_size;
  const float* feats     = (const float*)d_in[0];
  const float* sent_feat = (const float*)d_in[1];
  const float* sent_iou  = (const float*)d_in[2];
  const float* sent_fus  = (const float*)d_in[3];
  const float* W1        = (const float*)d_in[4];
  const float* b1        = (const float*)d_in[5];
  const float* W2        = (const float*)d_in[6];
  const float* b2        = (const float*)d_in[7];
  const float* Wp        = (const float*)d_in[8];
  const float* Wiou      = (const float*)d_in[9];
  const float* Wfuse     = (const float*)d_in[10];
  float* out = (float*)d_out;

  const size_t lds1 = 66560 + 16384 + 2048 + 4608;        // 89600 B
  const size_t lds2 = 4096 + 33792 + 17408;               // 55296 B
  const size_t packBytes = (size_t)512 * 512 * 2;         // 512 KB bf16 packed W

  if (ws_size >= packBytes) {
    __bf16* Apk = (__bf16*)d_ws;
    mmn_packw_kernel<<<dim3(512), 256, 0, stream>>>(Wiou, Wp, Apk);
    mmn_map2d_kernel<true><<<dim3(Nv, Bv), 256, lds1, stream>>>(
        feats, sent_feat, sent_iou, Wp, Wiou, Apk, out);
  } else {
    mmn_map2d_kernel<false><<<dim3(Nv, Bv), 256, lds1, stream>>>(
        feats, sent_feat, sent_iou, Wp, Wiou, (const __bf16*)nullptr, out);
  }
  mmn_fusion_kernel<<<dim3(Bv), 256, lds2, stream>>>(
      feats, sent_fus, W1, b1, W2, b2, Wfuse, out);
}
